// AFNO2D_27865747816696
// MI455X (gfx1250) — matrix operand order repaired and checked by reading, not compiled, not run on hardware
//
#include <hip/hip_runtime.h>

// ---------------------------------------------------------------------------
// AFNO2D for MI455X (gfx1250): rfft2 -> block-diag complex MLP (bf16 WMMA,
// f32 accum) -> irfft2 + residual.
// ---------------------------------------------------------------------------

#define HG 64
#define WG 64
#define WF 33          // WG/2+1
#define NB 8
#define BS 128
#define BDIM 8
#define CH (NB*BS)     // 1024
#define MTOT (BDIM*HG*WF)  // 16896 spectral positions
#define LAMBDA 0.01f

typedef __attribute__((ext_vector_type(16))) __bf16        v16bf;
typedef __attribute__((ext_vector_type(8)))  float         v8f;
typedef __attribute__((ext_vector_type(8)))  unsigned int  v8u;

// ----- helpers --------------------------------------------------------------

__device__ __forceinline__ unsigned int pack_bf16x2(float a, float b) {
    union { float f; unsigned int u; } ua, ub;
    ua.f = a; ub.f = b;
    unsigned int x = ua.u + (0x7FFFu + ((ua.u >> 16) & 1u));   // RNE
    unsigned int y = ub.u + (0x7FFFu + ((ub.u >> 16) & 1u));
    return ((x >> 16) & 0xFFFFu) | (y & 0xFFFF0000u);
}

__device__ __forceinline__ v8f wmma_bf16(const unsigned int* a,
                                         const unsigned int* b, v8f c) {
    v8u av, bv;
#pragma unroll
    for (int i = 0; i < 8; ++i) { av[i] = a[i]; bv[i] = b[i]; }
    return __builtin_amdgcn_wmma_f32_16x16x32_bf16(
        false, __builtin_bit_cast(v16bf, av),
        false, __builtin_bit_cast(v16bf, bv),
        (short)0, c, false, false);
}

__device__ __forceinline__ float2 cadd(float2 a, float2 b) { return make_float2(a.x + b.x, a.y + b.y); }
__device__ __forceinline__ float2 csub(float2 a, float2 b) { return make_float2(a.x - b.x, a.y - b.y); }
__device__ __forceinline__ float2 cmul(float2 a, float2 b) {
    return make_float2(a.x * b.x - a.y * b.y, a.x * b.y + a.y * b.x);
}

// fully-unrolled in-register 64-point complex FFT (DIT radix-2, e^{-i...})
__device__ __forceinline__ void fft64(float2* v) {
#pragma unroll
    for (int i = 0; i < 64; ++i) {
        int j = ((i & 1) << 5) | ((i & 2) << 3) | ((i & 4) << 1) |
                ((i & 8) >> 1) | ((i & 16) >> 3) | ((i & 32) >> 5);
        if (j > i) { float2 t = v[i]; v[i] = v[j]; v[j] = t; }
    }
    float2 tw[32];
#pragma unroll
    for (int k = 0; k < 32; ++k) {
        float a = -3.14159265358979323846f * (float)k / 32.0f;
        tw[k] = make_float2(__cosf(a), __sinf(a));
    }
#pragma unroll
    for (int s = 1; s <= 6; ++s) {
        const int m = 1 << s, half = m >> 1, tstep = 64 >> s;
#pragma unroll
        for (int g = 0; g < 64; g += m) {
#pragma unroll
            for (int k = 0; k < half; ++k) {
                float2 w = tw[k * tstep];
                float2 t = cmul(w, v[g + k + half]);
                float2 u = v[g + k];
                v[g + k] = cadd(u, t);
                v[g + k + half] = csub(u, t);
            }
        }
    }
}

// ----- stage A: real FFT over W, ortho scale folded here (1/64 total) ------

__global__ __launch_bounds__(256) void fft_w_fwd(const float* __restrict__ x,
                                                 float* __restrict__ sr,
                                                 float* __restrict__ si) {
    int tid = blockIdx.x * 256 + threadIdx.x;       // BDIM*HG*CH threads
    int c = tid & (CH - 1);
    int line = tid >> 10;                           // b*HG + h
    const float* xp = x + (size_t)line * WG * CH + c;
    float2 v[64];
#pragma unroll
    for (int w = 0; w < 64; ++w) v[w] = make_float2(xp[(size_t)w * CH], 0.f);
    fft64(v);
    float* srp = sr + (size_t)line * WF * CH + c;
    float* sip = si + (size_t)line * WF * CH + c;
    const float scale = 1.0f / 64.0f;               // full 2D ortho factor
#pragma unroll
    for (int w = 0; w < WF; ++w) {
        srp[(size_t)w * CH] = v[w].x * scale;
        sip[(size_t)w * CH] = v[w].y * scale;
    }
}

// ----- stage B / D1: complex FFT over H (conj flag -> unnormalized IDFT) ---

__global__ __launch_bounds__(256) void fft_h_kernel(float* __restrict__ sr,
                                                    float* __restrict__ si,
                                                    int conj) {
    int tid = blockIdx.x * 256 + threadIdx.x;       // BDIM*WF*CH threads
    int c = tid & (CH - 1);
    int line = tid >> 10;                           // b*WF + w
    int b = line / WF;
    int w = line - b * WF;
    size_t base = ((size_t)b * HG * WF + w) * CH + c;
    const size_t stride = (size_t)WF * CH;
    const float sgn = conj ? -1.f : 1.f;
    float2 v[64];
#pragma unroll
    for (int h = 0; h < 64; ++h)
        v[h] = make_float2(sr[base + h * stride], sgn * si[base + h * stride]);
    fft64(v);
#pragma unroll
    for (int h = 0; h < 64; ++h) {
        sr[base + h * stride] = v[h].x;
        si[base + h * stride] = sgn * v[h].y;
    }
}

// ----- weight packing into per-lane WMMA fragment order --------------------
// mats: 0=W1r(A) 1=W1i(A) 2=-W1i(A) 3=W2r(B) 4=W2i(B) 5=-W2i(B)
// pw dword index = ((((mat*8+nb)*8 + t)*4 + kc)*32 + lane)*8 + j
#define PW_DWORDS (6 * 8 * 8 * 4 * 256)

__global__ __launch_bounds__(256) void pack_weights(const float* __restrict__ w1,
                                                    const float* __restrict__ w2,
                                                    unsigned int* __restrict__ pw) {
    int tid = blockIdx.x * 256 + threadIdx.x;
    if (tid >= PW_DWORDS) return;
    int j    = tid & 7;
    int lane = (tid >> 3) & 31;
    int kc   = (tid >> 8) & 3;
    int t    = (tid >> 10) & 7;
    int nb   = (tid >> 13) & 7;
    int mat  = tid >> 16;
    int laneHi = lane >> 4;
    int o = t * 16 + (lane & 15);
    int i;
    if (mat < 3)  // A-operand layout (16x32 bf16): K split {0-7,16-23} / +8
        i = kc * 32 + ((j < 4) ? 2 * j : 8 + 2 * j) + (laneHi ? 8 : 0);
    else          // B-operand layout (32x16 bf16): K split {0-7,16-23} / +8
        i = kc * 32 + ((j < 4) ? 2 * j : 8 + 2 * j) + (laneHi ? 8 : 0);
    const float* W;
    float s = 1.f;
    switch (mat) {
        case 0: W = w1 + (size_t)nb * BS * BS;            break;
        case 1: W = w1 + (size_t)(8 + nb) * BS * BS;      break;
        case 2: W = w1 + (size_t)(8 + nb) * BS * BS; s = -1.f; break;
        case 3: W = w2 + (size_t)nb * BS * BS;            break;
        case 4: W = w2 + (size_t)(8 + nb) * BS * BS;      break;
        default:W = w2 + (size_t)(8 + nb) * BS * BS; s = -1.f; break;
    }
    float a = s * W[(size_t)i * BS + o];
    float b = s * W[(size_t)(i + 1) * BS + o];
    pw[tid] = pack_bf16x2(a, b);
}

#define FRAG(mat, nb, t, kc) ((unsigned)((((((mat)*8 + (nb))*8 + (t))*4 + (kc)) << 8)))

// ----- stage C: block-diagonal complex MLP, bf16 WMMA, in-place ------------
// one wave per (16-position tile, block). layer1 computed transposed so its
// D fragments repack in-register into layer2 A fragments.

__global__ __launch_bounds__(256) void block_mlp(float* __restrict__ sr,
                                                 float* __restrict__ si,
                                                 const unsigned int* __restrict__ pw,
                                                 const float* __restrict__ b1,
                                                 const float* __restrict__ b2) {
    const int lane   = threadIdx.x & 31;
    const int wid    = threadIdx.x >> 5;
    const int nb     = wid;                 // 8 waves/block = 8 feature blocks
    const int mtile  = blockIdx.x;          // 1056 tiles of 16 positions
    const int laneHi = lane >> 4;
    const int ln     = lane & 15;

    // X^T as B fragments (K = input feature i)
    unsigned int xr[4][8], xi[4][8];
    {
        const int m = mtile * 16 + ln;
        const float* pr = sr + (size_t)m * CH + nb * BS;
        const float* pi = si + (size_t)m * CH + nb * BS;
#pragma unroll
        for (int kc = 0; kc < 4; ++kc)
#pragma unroll
            for (int j = 0; j < 8; ++j) {
                int i = kc * 32 + ((j < 4) ? 2 * j : 8 + 2 * j) + (laneHi ? 8 : 0);
                xr[kc][j] = pack_bf16x2(pr[i], pr[i + 1]);
                xi[kc][j] = pack_bf16x2(pi[i], pi[i + 1]);
            }
    }

    const float* b1r = b1 + nb * BS;
    const float* b1i = b1 + (8 + nb) * BS;
    unsigned int a2r[4][8], a2i[4][8];      // layer-2 A fragments (o1 bf16)

    // ---- layer 1: o1^T = relu(W1r^T Xr^T - W1i^T Xi^T + b1r) etc. ----
#pragma unroll
    for (int t = 0; t < 8; ++t) {
        v8f accr, acci;
#pragma unroll
        for (int j = 0; j < 8; ++j) { accr[j] = 0.f; acci[j] = 0.f; }
        unsigned int wf[8];
#pragma unroll
        for (int kc = 0; kc < 4; ++kc) {
            const unsigned int* f0 = pw + FRAG(0, nb, t, kc) + lane * 8; // W1r
#pragma unroll
            for (int q = 0; q < 8; ++q) wf[q] = f0[q];
            accr = wmma_bf16(wf, xr[kc], accr);
            acci = wmma_bf16(wf, xi[kc], acci);
            const unsigned int* f2 = pw + FRAG(2, nb, t, kc) + lane * 8; // -W1i
#pragma unroll
            for (int q = 0; q < 8; ++q) wf[q] = f2[q];
            accr = wmma_bf16(wf, xi[kc], accr);
            const unsigned int* f1 = pw + FRAG(1, nb, t, kc) + lane * 8; // W1i
#pragma unroll
            for (int q = 0; q < 8; ++q) wf[q] = f1[q];
            acci = wmma_bf16(wf, xr[kc], acci);
        }
        // bias + relu; D rows here are output features o = t*16 + laneHi*8 + j
        const int ob = t * 16 + laneHi * 8;
#pragma unroll
        for (int j = 0; j < 8; ++j) {
            float r = accr[j] + b1r[ob + j]; accr[j] = r > 0.f ? r : 0.f;
            float q = acci[j] + b1i[ob + j]; acci[j] = q > 0.f ? q : 0.f;
        }
        // in-register repack: D(tile t) -> layer2 A fragment chunk t/2
        const int kc2 = t >> 1, hb = (t & 1) * 4;
#pragma unroll
        for (int j2 = 0; j2 < 4; ++j2) {
            a2r[kc2][hb + j2] = pack_bf16x2(accr[2 * j2], accr[2 * j2 + 1]);
            a2i[kc2][hb + j2] = pack_bf16x2(acci[2 * j2], acci[2 * j2 + 1]);
        }
    }

    // ---- layer 2: o2 = o1*W2r -/+ ... + b2, softshrink, store in-place ----
    const float* b2r = b2 + nb * BS;
    const float* b2i = b2 + (8 + nb) * BS;
#pragma unroll
    for (int t = 0; t < 8; ++t) {
        v8f cr, ci;
#pragma unroll
        for (int j = 0; j < 8; ++j) { cr[j] = 0.f; ci[j] = 0.f; }
        unsigned int wf[8];
#pragma unroll
        for (int kc = 0; kc < 4; ++kc) {
            const unsigned int* f3 = pw + FRAG(3, nb, t, kc) + lane * 8; // W2r
#pragma unroll
            for (int q = 0; q < 8; ++q) wf[q] = f3[q];
            cr = wmma_bf16(a2r[kc], wf, cr);
            ci = wmma_bf16(a2i[kc], wf, ci);
            const unsigned int* f5 = pw + FRAG(5, nb, t, kc) + lane * 8; // -W2i
#pragma unroll
            for (int q = 0; q < 8; ++q) wf[q] = f5[q];
            cr = wmma_bf16(a2i[kc], wf, cr);
            const unsigned int* f4 = pw + FRAG(4, nb, t, kc) + lane * 8; // W2i
#pragma unroll
            for (int q = 0; q < 8; ++q) wf[q] = f4[q];
            ci = wmma_bf16(a2r[kc], wf, ci);
        }
        const int o = t * 16 + ln;                 // D column = output feature
        const float br = b2r[o], bi = b2i[o];
        const int mrow0 = mtile * 16 + laneHi * 8; // D rows = positions
        float* yrp = sr + (size_t)mrow0 * CH + nb * BS + o;
        float* yip = si + (size_t)mrow0 * CH + nb * BS + o;
#pragma unroll
        for (int j = 0; j < 8; ++j) {
            float r = cr[j] + br;
            r = (r > LAMBDA) ? r - LAMBDA : ((r < -LAMBDA) ? r + LAMBDA : 0.f);
            float q = ci[j] + bi;
            q = (q > LAMBDA) ? q - LAMBDA : ((q < -LAMBDA) ? q + LAMBDA : 0.f);
            yrp[(size_t)j * CH] = r;
            yip[(size_t)j * CH] = q;
        }
    }
}

// ----- stage D2: hermitian inverse FFT over W, scale, residual add ---------

__global__ __launch_bounds__(256) void ifft_w_out(const float* __restrict__ yr,
                                                  const float* __restrict__ yi,
                                                  const float* __restrict__ x,
                                                  float* __restrict__ out) {
    int tid = blockIdx.x * 256 + threadIdx.x;       // BDIM*HG*CH threads
    int c = tid & (CH - 1);
    int line = tid >> 10;                           // b*HG + h
    size_t base = (size_t)line * WF * CH + c;
    float2 v[64];
#pragma unroll
    for (int w = 0; w < WF; ++w)                    // u = conj(Y)
        v[w] = make_float2(yr[base + (size_t)w * CH], -yi[base + (size_t)w * CH]);
#pragma unroll
    for (int w = WF; w < 64; ++w)                   // u[w] = Y[64-w]
        v[w] = make_float2(v[64 - w].x, -v[64 - w].y);
    fft64(v);                                       // Re(fft(conj Y)) = IDFT
    const float scale = 1.0f / 64.0f;               // inverse 2D ortho factor
    size_t obase = (size_t)line * WG * CH + c;
#pragma unroll
    for (int w = 0; w < 64; ++w)
        out[obase + (size_t)w * CH] = v[w].x * scale + x[obase + (size_t)w * CH];
}

// ---------------------------------------------------------------------------

extern "C" void kernel_launch(void* const* d_in, const int* in_sizes, int n_in,
                              void* d_out, int out_size, void* d_ws, size_t ws_size,
                              hipStream_t stream) {
    const float* x  = (const float*)d_in[0];
    const float* w1 = (const float*)d_in[1];
    const float* b1 = (const float*)d_in[2];
    const float* w2 = (const float*)d_in[3];
    const float* b2 = (const float*)d_in[4];
    float* out = (float*)d_out;

    char* ws = (char*)d_ws;
    unsigned int* pw = (unsigned int*)ws;                    // 1.5 MB packed weights
    const size_t NS = (size_t)BDIM * HG * WF * CH;           // 17,301,504 floats
    float* sr = (float*)(ws + (size_t)(2u << 20));
    float* si = sr + NS;

    pack_weights<<<PW_DWORDS / 256, 256, 0, stream>>>(w1, w2, pw);
    fft_w_fwd  <<<(BDIM * HG * CH) / 256, 256, 0, stream>>>(x, sr, si);
    fft_h_kernel<<<(BDIM * WF * CH) / 256, 256, 0, stream>>>(sr, si, 0);
    block_mlp  <<<MTOT / 16, 256, 0, stream>>>(sr, si, pw, b1, b2);
    fft_h_kernel<<<(BDIM * WF * CH) / 256, 256, 0, stream>>>(sr, si, 1);
    ifft_w_out <<<(BDIM * HG * CH) / 256, 256, 0, stream>>>(sr, si, x, out);
}
